// MultiHeadAttention_57294863729465
// MI455X (gfx1250) — compile-verified, matrix operand order checked by reading, not run on hardware
//
#include <hip/hip_runtime.h>
#include <hip/hip_bf16.h>
#include <stdint.h>

// ---------------------------------------------------------------------------
// MHA forward for MI455X (gfx1250, wave32, WMMA 16x16x32 f16).
// Pipeline: weight convert/transpose (f32->f16) -> QKV GEMMs (WMMA, f16 out,
//           Q pre-scaled by 1/sqrt(dk)) -> V transpose to [b,h,dk,s] ->
//           flash attention (WMMA + online softmax) -> out projection (f32).
// Tile staging uses CDNA5 async copies (global_load_async_to_lds_b128 +
// s_wait_asynccnt) with double-buffered LDS so global->LDS traffic overlaps
// the WMMA stream.
// ---------------------------------------------------------------------------

typedef __attribute__((ext_vector_type(16))) _Float16 v16h;
typedef __attribute__((ext_vector_type(8)))  float    v8f;
typedef __attribute__((ext_vector_type(8)))  _Float16 f16x8;
typedef __attribute__((ext_vector_type(4)))  _Float16 f16x4;

#define SEQ    2048
#define DMODEL 1024
#define NHEAD  16
#define DK     64
#define NB     4
#define NTOK   (NB * SEQ)      // 8192 rows

// CDNA5 async global->LDS copy, 16B per lane. Generic pointers to LDS carry
// the LDS byte offset in their low 32 bits (flat aperture rule), which is
// exactly the VDST operand of GLOBAL_LOAD_ASYNC_TO_LDS. Tracked by ASYNCcnt.
__device__ __forceinline__ void async_cp16(void* lds_dst, const void* gsrc) {
  unsigned l = (unsigned)(uintptr_t)lds_dst;
  unsigned long long g = (unsigned long long)(uintptr_t)gsrc;
  asm volatile("global_load_async_to_lds_b128 %0, %1, off"
               :: "v"(l), "v"(g) : "memory");
}
__device__ __forceinline__ void wait_async0() {
  asm volatile("s_wait_asynccnt 0x0" ::: "memory");
}

// Load one WMMA fragment from an LDS row (16-bit A/B layout, wave32):
//   lanes 0-15: K koff..koff+7 and koff+16..koff+23 ; lanes 16-31: +8.
// Caller passes koff = kstep*32 + (lane>=16 ? 8 : 0).
__device__ __forceinline__ v16h ld_frag(const _Float16* rowbase, int koff) {
  f16x8 lo = *reinterpret_cast<const f16x8*>(rowbase + koff);
  f16x8 hi = *reinterpret_cast<const f16x8*>(rowbase + koff + 16);
  v16h r;
#pragma unroll
  for (int i = 0; i < 8; ++i) { r[i] = lo[i]; r[i + 8] = hi[i]; }
  return r;
}

// ---------------------------------------------------------------------------
// Weight convert + transpose: W fp32 [k][n] -> Wt f16 [n][k]
// ---------------------------------------------------------------------------
__global__ __launch_bounds__(256) void wconv_kernel(const float* __restrict__ W,
                                                    _Float16* __restrict__ Wt) {
  int id = blockIdx.x * 256 + threadIdx.x;      // 1M elements
  int n = id >> 10;
  int k = id & 1023;
  Wt[(size_t)n * DMODEL + k] = (_Float16)W[(size_t)k * DMODEL + n];
}

// ---------------------------------------------------------------------------
// V transpose: V16 [token][d] -> Vt16 [(b*16+h)*64+dk][s]   (f16)
// ---------------------------------------------------------------------------
__global__ __launch_bounds__(256) void vtrans_kernel(const _Float16* __restrict__ V,
                                                     _Float16* __restrict__ Vt) {
  int id = blockIdx.x * 256 + threadIdx.x;      // 8192*1024/8 = 1M
  int n  = id >> 7;                             // token row
  int dc = (id & 127) * 8;                      // d chunk
  f16x8 v = *reinterpret_cast<const f16x8*>(V + (size_t)n * DMODEL + dc);
  int b = n >> 11, s = n & (SEQ - 1);
#pragma unroll
  for (int j = 0; j < 8; ++j) {
    int d = dc + j, h = d >> 6, dk = d & 63;
    Vt[(((size_t)(b * NHEAD + h)) * DK + dk) * SEQ + s] = v[j];
  }
}

// ---------------------------------------------------------------------------
// WMMA GEMM: C[8192][1024] = (A[8192][1024] @ W + bias) * scale
// Bt is W transposed: Bt[n][k] f16. Block tile 128x128, BK=32, 8 waves
// (4x2), each wave 32x64 = 2x4 v_wmma_f32_16x16x32_f16. Double-buffered LDS;
// B (and A when f16) staged with async copies, fp32 A prefetched to regs and
// converted behind the WMMAs.
// ---------------------------------------------------------------------------
template <bool A_F16, bool OUT_F16>
__global__ __launch_bounds__(256) void gemm_wmma(const void* __restrict__ Ap,
                                                 const _Float16* __restrict__ Bt,
                                                 const float* __restrict__ bias,
                                                 float scale,
                                                 void* __restrict__ Cp) {
  __shared__ __align__(16) _Float16 As[2][128][40];   // 10240 B each
  __shared__ __align__(16) _Float16 Bs[2][128][40];

  const int tid  = threadIdx.x;
  const int lane = tid & 31, wid = tid >> 5;
  const int ln   = lane & 15, half = lane >> 4;
  const int wm   = wid & 3,  wn  = wid >> 2;          // 4x2 wave grid
  const int m0   = blockIdx.y * 128;
  const int n0   = blockIdx.x * 128;

  auto issueB = [&](int k0, int buf) {
#pragma unroll
    for (int r = 0; r < 2; ++r) {
      int c = tid + r * 256;
      int row = c >> 2, kc = (c & 3) * 8;
      async_cp16(&Bs[buf][row][kc], Bt + (size_t)(n0 + row) * DMODEL + k0 + kc);
    }
  };
  auto issueA16 = [&](int k0, int buf) {
    const _Float16* Ah = (const _Float16*)Ap;
#pragma unroll
    for (int r = 0; r < 2; ++r) {
      int c = tid + r * 256;
      int row = c >> 2, kc = (c & 3) * 8;
      async_cp16(&As[buf][row][kc], Ah + (size_t)(m0 + row) * DMODEL + k0 + kc);
    }
  };
  auto loadA32 = [&](int k0, float4* av) {
    const float* Af = (const float*)Ap;
#pragma unroll
    for (int r = 0; r < 4; ++r) {
      int f = tid + r * 256;
      int row = f >> 3, kc = (f & 7) * 4;
      av[r] = *reinterpret_cast<const float4*>(Af + (size_t)(m0 + row) * DMODEL + k0 + kc);
    }
  };
  auto storeA32 = [&](int buf, const float4* av) {
#pragma unroll
    for (int r = 0; r < 4; ++r) {
      int f = tid + r * 256;
      int row = f >> 3, kc = (f & 7) * 4;
      f16x4 h4;
      h4[0] = (_Float16)av[r].x; h4[1] = (_Float16)av[r].y;
      h4[2] = (_Float16)av[r].z; h4[3] = (_Float16)av[r].w;
      *reinterpret_cast<f16x4*>(&As[buf][row][kc]) = h4;
    }
  };

  v8f acc[2][4];
#pragma unroll
  for (int mt = 0; mt < 2; ++mt)
#pragma unroll
    for (int t = 0; t < 4; ++t) {
      float bv = bias[n0 + wn * 64 + t * 16 + ln];
#pragma unroll
      for (int i = 0; i < 8; ++i) acc[mt][t][i] = bv;
    }

  // prologue: stage k-chunk 0 into buffer 0
  float4 pav[4];
  if (A_F16) {
    issueA16(0, 0);
  } else {
    loadA32(0, pav);
    storeA32(0, pav);
  }
  issueB(0, 0);

  int buf = 0;
  for (int i = 0; i < DMODEL / 32; ++i) {
    wait_async0();
    __syncthreads();                       // tile i visible to all waves
    const int nk = (i + 1) * 32;
    float4 nv[4];
    if (nk < DMODEL) {                     // stage tile i+1 into other buffer
      issueB(nk, buf ^ 1);
      if (A_F16) issueA16(nk, buf ^ 1);
      else       loadA32(nk, nv);
    }

    v16h af[2];
#pragma unroll
    for (int mt = 0; mt < 2; ++mt)
      af[mt] = ld_frag(&As[buf][wm * 32 + mt * 16 + ln][0], half * 8);
#pragma unroll
    for (int t = 0; t < 4; ++t) {
      v16h bf = ld_frag(&Bs[buf][wn * 64 + t * 16 + ln][0], half * 8);
#pragma unroll
      for (int mt = 0; mt < 2; ++mt)
        acc[mt][t] = __builtin_amdgcn_wmma_f32_16x16x32_f16(
            false, af[mt], false, bf, (short)0, acc[mt][t], false, false);
    }

    if (!A_F16 && nk < DMODEL) storeA32(buf ^ 1, nv);
    buf ^= 1;
  }

#pragma unroll
  for (int mt = 0; mt < 2; ++mt)
#pragma unroll
    for (int t = 0; t < 4; ++t) {
      int n  = n0 + wn * 64 + t * 16 + ln;
      int mb = m0 + wm * 32 + mt * 16 + half * 8;
#pragma unroll
      for (int i = 0; i < 8; ++i) {
        size_t idx = (size_t)(mb + i) * DMODEL + n;
        float v = acc[mt][t][i] * scale;
        if (OUT_F16) ((_Float16*)Cp)[idx] = (_Float16)v;
        else         ((float*)Cp)[idx]    = v;
      }
    }
}

// ---------------------------------------------------------------------------
// Flash attention: per (b,h) and 128-query block. Q pre-scaled by 1/sqrt(dk).
// K/V tiles double-buffered via async copies; online softmax with 16-lane
// shuffle reductions; P converted C-layout -> A-layout via per-wave LDS.
// Shared arena (55296 B): Q staging tile overlays the K/V buffers (Q
// fragments live in registers after the prologue).
// ---------------------------------------------------------------------------
__global__ __launch_bounds__(256) void attn_kernel(const _Float16* __restrict__ Q,
                                                   const _Float16* __restrict__ K,
                                                   const _Float16* __restrict__ Vt,
                                                   const int* __restrict__ mask,
                                                   _Float16* __restrict__ Ctx) {
  __shared__ __align__(16) unsigned char arena[55296];
  // layout: Ks[0] @0, Ks[1] @9216, Vs[0] @18432, Vs[1] @27648, Ps @36864
  // Q staging tile (128x72 f16 = 18432 B) overlays @0 during the prologue.
  auto KsBuf = [&](int b) { return (_Float16(*)[72])(arena + (size_t)b * 9216); };
  auto VsBuf = [&](int b) { return (_Float16(*)[72])(arena + 18432 + (size_t)b * 9216); };
  _Float16 (*Qs)[72] = (_Float16(*)[72])(arena);
  _Float16 (*Ps)[72] = (_Float16(*)[72])(arena + 36864);   // rows = wid*16 + r

  const int tid  = threadIdx.x;
  const int lane = tid & 31, wid = tid >> 5;
  const int ln   = lane & 15, half = lane >> 4;
  const int q0   = blockIdx.x * 128;
  const int bh   = blockIdx.y;
  const int b    = bh >> 4, h = bh & 15;
  const float NEG_INF = -__builtin_inff();

  // ---- prologue: async-stage Q tile, pull fragments to registers ----
#pragma unroll
  for (int r = 0; r < 4; ++r) {
    int c = tid + r * 256;
    int row = c >> 3, kc = (c & 7) * 8;
    async_cp16(&Qs[row][kc],
               Q + (size_t)(b * SEQ + q0 + row) * DMODEL + h * DK + kc);
  }
  wait_async0();
  __syncthreads();

  v16h qa[2];
#pragma unroll
  for (int kc = 0; kc < 2; ++kc)
    qa[kc] = ld_frag(&Qs[wid * 16 + ln][0], kc * 32 + half * 8);
  __syncthreads();                     // arena reuse safe: qa in registers

  auto issueKV = [&](int kb, int bsel) {
    const int kb0 = kb * 64;
    _Float16 (*Ks)[72] = KsBuf(bsel);
    _Float16 (*Vs)[72] = VsBuf(bsel);
#pragma unroll
    for (int r = 0; r < 2; ++r) {
      int c = tid + r * 256;
      int row = c >> 3, kc = (c & 7) * 8;
      async_cp16(&Ks[row][kc],
                 K + (size_t)(b * SEQ + kb0 + row) * DMODEL + h * DK + kc);
      async_cp16(&Vs[row][kc],
                 Vt + ((size_t)(bh * DK + row)) * SEQ + kb0 + kc);
    }
  };
  issueKV(0, 0);

  v8f o[4];
  float mrow[8], lrow[8], corr[8];
#pragma unroll
  for (int t = 0; t < 4; ++t)
#pragma unroll
    for (int i = 0; i < 8; ++i) o[t][i] = 0.0f;
#pragma unroll
  for (int i = 0; i < 8; ++i) { mrow[i] = NEG_INF; lrow[i] = 0.0f; }

  for (int kb = 0; kb < SEQ / 64; ++kb) {
    wait_async0();
    __syncthreads();                   // tile kb resident; prev reads done
    if (kb + 1 < SEQ / 64) issueKV(kb + 1, (kb + 1) & 1);

    _Float16 (*Ks)[72] = KsBuf(kb & 1);
    _Float16 (*Vs)[72] = VsBuf(kb & 1);
    const int kb0 = kb * 64;

    // S = Qs @ K^T over dk=64 (two WMMA k-steps)
    v8f sc[4];
#pragma unroll
    for (int t = 0; t < 4; ++t) {
      v16h b0 = ld_frag(&Ks[t * 16 + ln][0], half * 8);
      v16h b1 = ld_frag(&Ks[t * 16 + ln][0], 32 + half * 8);
      v8f z = {};
      sc[t] = __builtin_amdgcn_wmma_f32_16x16x32_f16(false, qa[0], false, b0, (short)0, z, false, false);
      sc[t] = __builtin_amdgcn_wmma_f32_16x16x32_f16(false, qa[1], false, b1, (short)0, sc[t], false, false);
    }
    // mask (B,1,1,S): masked key column -> -inf
#pragma unroll
    for (int t = 0; t < 4; ++t) {
      int kg = kb0 + t * 16 + ln;
      if (mask[(size_t)b * SEQ + kg] == 0) {
#pragma unroll
        for (int i = 0; i < 8; ++i) sc[t][i] = NEG_INF;
      }
    }
    // online softmax: C-fragment rows live per half-wave, cols across 16 lanes
#pragma unroll
    for (int i = 0; i < 8; ++i) {
      float r = fmaxf(fmaxf(sc[0][i], sc[1][i]), fmaxf(sc[2][i], sc[3][i]));
#pragma unroll
      for (int msk = 1; msk < 16; msk <<= 1) r = fmaxf(r, __shfl_xor(r, msk, 32));
      float mn = fmaxf(mrow[i], r);
      corr[i]  = __expf(mrow[i] - mn);
      mrow[i]  = mn;
    }
#pragma unroll
    for (int t = 0; t < 4; ++t)
#pragma unroll
      for (int i = 0; i < 8; ++i) sc[t][i] = __expf(sc[t][i] - mrow[i]);
#pragma unroll
    for (int i = 0; i < 8; ++i) {
      float rs = sc[0][i] + sc[1][i] + sc[2][i] + sc[3][i];
#pragma unroll
      for (int msk = 1; msk < 16; msk <<= 1) rs += __shfl_xor(rs, msk, 32);
      lrow[i] = lrow[i] * corr[i] + rs;
#pragma unroll
      for (int t = 0; t < 4; ++t) o[t][i] *= corr[i];
    }
    // P: C-layout -> per-wave LDS -> A-layout
#pragma unroll
    for (int t = 0; t < 4; ++t)
#pragma unroll
      for (int i = 0; i < 8; ++i)
        Ps[wid * 16 + half * 8 + i][t * 16 + ln] = (_Float16)sc[t][i];
    __syncthreads();
    // O += P @ V (inner = 64 keys, two WMMA k-steps)
#pragma unroll
    for (int kc = 0; kc < 2; ++kc) {
      v16h pa = ld_frag(&Ps[wid * 16 + ln][0], kc * 32 + half * 8);
#pragma unroll
      for (int t = 0; t < 4; ++t) {
        v16h vb = ld_frag(&Vs[t * 16 + ln][0], kc * 32 + half * 8);
        o[t] = __builtin_amdgcn_wmma_f32_16x16x32_f16(false, pa, false, vb, (short)0, o[t], false, false);
      }
    }
  }

  // normalize and write context (f16, [token][d] row-major)
#pragma unroll
  for (int t = 0; t < 4; ++t)
#pragma unroll
    for (int i = 0; i < 8; ++i) {
      int r = half * 8 + i;
      float val = o[t][i] / lrow[i];
      size_t idx = (size_t)(b * SEQ + q0 + wid * 16 + r) * DMODEL + h * DK + t * 16 + ln;
      Ctx[idx] = (_Float16)val;
    }
}

// ---------------------------------------------------------------------------

extern "C" void kernel_launch(void* const* d_in, const int* in_sizes, int n_in,
                              void* d_out, int out_size, void* d_ws, size_t ws_size,
                              hipStream_t stream) {
  const float* q    = (const float*)d_in[0];
  const float* k    = (const float*)d_in[1];
  const float* v    = (const float*)d_in[2];
  const int*   mask = (const int*)d_in[3];
  const float* Wq   = (const float*)d_in[4];
  const float* bq   = (const float*)d_in[5];
  const float* Wk   = (const float*)d_in[6];
  const float* bk   = (const float*)d_in[7];
  const float* Wv   = (const float*)d_in[8];
  const float* bv   = (const float*)d_in[9];
  const float* Wo   = (const float*)d_in[10];
  const float* bo   = (const float*)d_in[11];

  char* ws = (char*)d_ws;
  size_t off = 0;
  auto carve = [&](size_t bytes) -> void* {
    void* p = ws + off;
    off += (bytes + 255) & ~(size_t)255;
    return p;
  };
  const size_t WBYTES = (size_t)DMODEL * DMODEL * sizeof(_Float16);      // 2 MiB
  const size_t TBYTES = (size_t)NTOK   * DMODEL * sizeof(_Float16);      // 16 MiB
  _Float16* WqT  = (_Float16*)carve(WBYTES);
  _Float16* WkT  = (_Float16*)carve(WBYTES);
  _Float16* WvT  = (_Float16*)carve(WBYTES);
  _Float16* WoT  = (_Float16*)carve(WBYTES);
  _Float16* Q16  = (_Float16*)carve(TBYTES);
  _Float16* K16  = (_Float16*)carve(TBYTES);
  _Float16* V16  = (_Float16*)carve(TBYTES);
  _Float16* Vt16 = (_Float16*)carve(TBYTES);
  _Float16* CTX16 = V16;   // reuse: V16 dead after vtrans

  dim3 blk(256);

  wconv_kernel<<<4096, blk, 0, stream>>>(Wq, WqT);
  wconv_kernel<<<4096, blk, 0, stream>>>(Wk, WkT);
  wconv_kernel<<<4096, blk, 0, stream>>>(Wv, WvT);
  wconv_kernel<<<4096, blk, 0, stream>>>(Wo, WoT);

  dim3 ggrid(DMODEL / 128, NTOK / 128);   // (8, 64)
  // Q projection folds the 1/sqrt(dk)=0.125 attention scale (exact pow2).
  gemm_wmma<false, true><<<ggrid, blk, 0, stream>>>(q, WqT, bq, 0.125f, Q16);
  gemm_wmma<false, true><<<ggrid, blk, 0, stream>>>(k, WkT, bk, 1.0f,   K16);
  gemm_wmma<false, true><<<ggrid, blk, 0, stream>>>(v, WvT, bv, 1.0f,   V16);

  vtrans_kernel<<<4096, blk, 0, stream>>>(V16, Vt16);

  dim3 agrid(SEQ / 128, NB * NHEAD);      // (16, 64)
  attn_kernel<<<agrid, blk, 0, stream>>>(Q16, K16, Vt16, mask, CTX16);

  gemm_wmma<true, false><<<ggrid, blk, 0, stream>>>(CTX16, WoT, bo, 1.0f, (float*)d_out);
}
